// Event_29291676958907
// MI455X (gfx1250) — matrix operand order repaired and checked by reading, not compiled, not run on hardware
//
#include <hip/hip_runtime.h>
#include <hip/hip_bf16.h>
#include <math.h>

// ---------------------------------------------------------------------------
// Problem constants (from reference)
// ---------------------------------------------------------------------------
#define BATCH       128
#define NSAMP       32768
#define NFREQ       128
#define IMP_SAMP    4096
#define NIMP        16
#define FSIZE       512
#define NFILT       16
#define HOPSZ       256

// Padded signal buffer for the big conv:
// x index range needed: [t0-2048, t0+2303], t0 max = 32512 -> len 36864 exactly
#define XPAD_LEN    36864
#define XPAD_FRONT  2048

typedef __attribute__((ext_vector_type(16))) __bf16 v16bf;
typedef __attribute__((ext_vector_type(8)))  float  v8f;

union Frag { v16bf v; uint4 q[2]; };

// ---------------------------------------------------------------------------
// JAX threefry2x32 noise, key = jax.random.key(42) -> (0, 42), shape 128*4096
// ---------------------------------------------------------------------------
__device__ __forceinline__ unsigned rotl32(unsigned x, unsigned r) {
    return (x << r) | (x >> (32u - r));
}

__device__ __forceinline__ float jax_noise(unsigned flat) {
    const unsigned half = (BATCH * IMP_SAMP) / 2u;  // 262144
    unsigned x0, x1; bool second;
    if (flat < half) { x0 = flat;        x1 = flat + half; second = false; }
    else             { x0 = flat - half; x1 = flat;        second = true;  }
    const unsigned k0 = 0u, k1 = 42u, k2 = 0x1BD11BDAu ^ k0 ^ k1;
    x0 += k0; x1 += k1;
#define TF_R4(a,b,c,d)                              \
    x0 += x1; x1 = rotl32(x1,a); x1 ^= x0;          \
    x0 += x1; x1 = rotl32(x1,b); x1 ^= x0;          \
    x0 += x1; x1 = rotl32(x1,c); x1 ^= x0;          \
    x0 += x1; x1 = rotl32(x1,d); x1 ^= x0;
    TF_R4(13,15,26,6)   x0 += k1; x1 += k2 + 1u;
    TF_R4(17,29,16,24)  x0 += k2; x1 += k0 + 2u;
    TF_R4(13,15,26,6)   x0 += k0; x1 += k1 + 3u;
    TF_R4(17,29,16,24)  x0 += k1; x1 += k2 + 4u;
    TF_R4(13,15,26,6)   x0 += k2; x1 += k0 + 5u;
#undef TF_R4
    unsigned bits = second ? x1 : x0;
    float u = __uint_as_float((bits >> 9) | 0x3f800000u) - 1.0f;  // [0,1)
    return fmaxf(-1.0f, u * 2.0f - 1.0f);                          // [-1,1)
}

// ---------------------------------------------------------------------------
// K1: balance -> bf16 row-major (A-matrix of GEMM1)
// ---------------------------------------------------------------------------
__global__ void k_bal(const float* __restrict__ balance, __bf16* __restrict__ abf) {
    int i = blockIdx.x * 256 + threadIdx.x;          // 128*128
    abf[i] = (__bf16)balance[i];
}

// ---------------------------------------------------------------------------
// K2: bank (128 x 32768 f32, row-major) -> bankT (32768 x 128 bf16) so that the
// WMMA B-matrix loads are K-contiguous per lane. LDS-tiled transpose.
// ---------------------------------------------------------------------------
__global__ void k_bankT(const float* __restrict__ bank, __bf16* __restrict__ bankT) {
    __shared__ float tile[32][33];
    int tx = threadIdx.x & 31, ty = threadIdx.x >> 5;     // 32 x 8
    int kt = blockIdx.x & 3;                               // 128/32
    int nt = blockIdx.x >> 2;                              // 32768/32
    int k0 = kt * 32, n0 = nt * 32;
#pragma unroll
    for (int i = 0; i < 4; i++) {
        int r = ty + i * 8;
        tile[r][tx] = bank[(size_t)(k0 + r) * NSAMP + n0 + tx];
    }
    __syncthreads();
#pragma unroll
    for (int i = 0; i < 4; i++) {
        int r = ty + i * 8;
        bankT[(size_t)(n0 + r) * NFREQ + k0 + tx] = (__bf16)tile[tx][r];
    }
}

// ---------------------------------------------------------------------------
// K3: zero the padded signal buffer (front/back zero pads for the big conv)
// ---------------------------------------------------------------------------
__global__ void k_zero(uint4* __restrict__ p) {
    uint4 z; z.x = z.y = z.z = z.w = 0u;
    p[(size_t)blockIdx.x * 256 + threadIdx.x] = z;
}

// ---------------------------------------------------------------------------
// K4: GEMM1 (res = balance @ bank) on bf16 WMMA, fused with the collapsed
// window/decay/overlap-add weighting; writes bf16 into xpad interior.
// One wave computes one 16x16 tile; K=128 -> 4 WMMA k-steps.
// ---------------------------------------------------------------------------
__global__ void k_gemm_res(const __bf16* __restrict__ abf,
                           const __bf16* __restrict__ bankT,
                           const float*  __restrict__ decay,
                           __bf16*       __restrict__ xpad) {
    int wid  = blockIdx.x * 8 + (threadIdx.x >> 5);   // 16384 waves
    int lane = threadIdx.x & 31;
    int mt = wid & 7;            // 8 M tiles (batch)
    int nt = wid >> 3;           // 2048 N tiles (samples)
    int h16 = lane >> 4, l15 = lane & 15;

    const __bf16* arow = abf   + (size_t)(mt * 16 + l15) * NFREQ;
    const __bf16* bcol = bankT + (size_t)(nt * 16 + l15) * NFREQ;

    v8f acc = {};
#pragma unroll
    for (int ks = 0; ks < 4; ks++) {
        Frag A, B;
        // A lane layout (16x32 bf16): elems 0..7 -> K = 8*h16+{0..7},
        //                             elems 8..15 -> K = 16+8*h16+{0..7}
        A.q[0] = *(const uint4*)(arow + ks * 32 + 8 * h16);
        A.q[1] = *(const uint4*)(arow + ks * 32 + 16 + 8 * h16);
        // B lane layout (32x16 bf16): n = lane&15, elems 0..7 -> K = 8*h16+{0..7}, elems 8..15 -> K = 16+8*h16+{0..7}
        B.q[0] = *(const uint4*)(bcol + ks * 32 + 8 * h16);
        B.q[1] = *(const uint4*)(bcol + ks * 32 + 16 + 8 * h16);
        acc = __builtin_amdgcn_wmma_f32_16x16x32_bf16(false, A.v, false, B.v,
                                                      (short)0, acc, false, false);
    }

    // Epilogue: res_sig[b,s] = res[b,s] * (ham[j]*d^(w0+1) + [s>=256]*ham[j+256]*d^w0)
    int s  = nt * 16 + l15;
    int w0 = s >> 8, j = s & 255;
    float hamA = 0.54f - 0.46f * cosf(6.283185307179586f * (float)j / 512.0f);
    float hamB = 0.54f - 0.46f * cosf(6.283185307179586f * (float)(j + 256) / 512.0f);
#pragma unroll
    for (int r = 0; r < 8; r++) {
        int b = mt * 16 + r + 8 * h16;                // C/D layout: row = r + 8*h16
        float d  = fminf(fmaxf(decay[b], 0.0f), 1.0f) + 1e-8f;
        float p0 = powf(d, (float)w0);
        float wt = hamA * (p0 * d) + ((s >= 256) ? hamB * p0 : 0.0f);
        xpad[(size_t)b * XPAD_LEN + XPAD_FRONT + s] = (__bf16)(acc[r] * wt);
    }
}

// ---------------------------------------------------------------------------
// K5: imp_sig = linear_resample(impulse_choice @ impulses^2, 4096) * noise
// ---------------------------------------------------------------------------
__global__ void k_impsig(const float* __restrict__ ic,
                         const float* __restrict__ impulses,
                         float* __restrict__ imp_sig) {
    unsigned flat = blockIdx.x * 256 + threadIdx.x;   // 128*4096
    unsigned b = flat >> 12;
    unsigned t = flat & 4095u;
    float pos = (t + 0.5f) * (1.0f / 32.0f) - 0.5f;   // scale = 128/4096
    pos = fminf(fmaxf(pos, 0.0f), 127.0f);
    int i0 = (int)floorf(pos);
    int i1 = min(i0 + 1, 127);
    float w = pos - (float)i0;
    float f0 = 0.f, f1 = 0.f;
#pragma unroll
    for (int j = 0; j < NIMP; j++) {
        float cj = ic[b * NIMP + j];
        float a0 = impulses[j * 128 + i0];
        float a1 = impulses[j * 128 + i1];
        f0 += cj * a0 * a0;
        f1 += cj * a1 * a1;
    }
    float env = f0 * (1.0f - w) + f1 * w;
    imp_sig[flat] = env * jax_noise(flat);
}

// ---------------------------------------------------------------------------
// K6: filt = (filter_choice @ filters) * hamming(512)
// ---------------------------------------------------------------------------
__global__ void k_filt(const float* __restrict__ fc,
                       const float* __restrict__ filters,
                       float* __restrict__ filt) {
    int idx = blockIdx.x * 256 + threadIdx.x;   // 128*512
    int c = idx >> 9, k = idx & 511;
    float s = 0.f;
#pragma unroll
    for (int j = 0; j < NFILT; j++) s += fc[c * NFILT + j] * filters[j * FSIZE + k];
    float ham = 0.54f - 0.46f * cosf(6.283185307179586f * (float)k / 512.0f);
    filt[idx] = s * ham;
}

// ---------------------------------------------------------------------------
// K7: impulse[c,t] = sum_k imp_sig[c, t+k-256] * filt[c,k]   (512-tap conv)
// LDS-staged: block = one channel x 256 outputs.
// ---------------------------------------------------------------------------
__global__ void k_impulse(const float* __restrict__ imp_sig,
                          const float* __restrict__ filt,
                          float* __restrict__ impulse) {
    __shared__ float ssig[768];
    __shared__ float sfil[512];
    int c     = blockIdx.x >> 4;
    int tbase = (blockIdx.x & 15) * 256;
    int tid   = threadIdx.x;
    for (int i = tid; i < 768; i += 256) {
        int t = tbase - 256 + i;
        ssig[i] = (t >= 0 && t < IMP_SAMP) ? imp_sig[c * IMP_SAMP + t] : 0.0f;
    }
    for (int i = tid; i < 512; i += 256) sfil[i] = filt[c * FSIZE + i];
    __syncthreads();
    float acc = 0.f;
#pragma unroll 8
    for (int k = 0; k < 512; k++) acc += ssig[tid + k] * sfil[k];
    impulse[c * IMP_SAMP + tbase + tid] = acc;
}

// ---------------------------------------------------------------------------
// K8: build banded-Toeplitz bf16 B tiles for the big conv, stored directly in
// WMMA lane layout:  btiles[c][kb][lane][e] = h[16*kb + u - n], u = e<8 ? 8*h16+e : 16+8*h16+(e-8)
// (zero outside the band 0 <= u-n < 16), n = lane&15, h16 = lane>>4.
// ---------------------------------------------------------------------------
__global__ void k_btiles(const float* __restrict__ impulse, __bf16* __restrict__ btiles) {
    unsigned idx  = blockIdx.x * 256 + threadIdx.x;   // c*256*32 + kb*32 + lane
    unsigned lane = idx & 31u;
    unsigned kb   = (idx >> 5) & 255u;
    unsigned c    = idx >> 13;
    int n = lane & 15, h16 = lane >> 4;
    union { __bf16 h[16]; uint4 q[2]; } P;
#pragma unroll
    for (int e = 0; e < 16; e++) {
        int d = (e < 8 ? 8 * h16 + e : 16 + 8 * h16 + (e - 8)) - n;
        float v = (d >= 0 && d < 16) ? impulse[c * IMP_SAMP + kb * 16 + d] : 0.0f;
        P.h[e] = (__bf16)v;
    }
    uint4* dst = (uint4*)(btiles + (size_t)idx * 16);
    dst[0] = P.q[0];
    dst[1] = P.q[1];
}

// ---------------------------------------------------------------------------
// K9: the big grouped conv on WMMA.
// y[t0+16m+n] = sum_kb sum_u  x[t0+16m+16kb-2048+u] * B_kb[u][n]
// One wave -> 256 consecutive outputs of one channel; 256 bf16 WMMA k-steps.
// A is half-reused between steps (one b128 load per step per lane).
// Epilogue fuses  (+ imp_padded) * amp  and writes d_out (f32).
// ---------------------------------------------------------------------------
__global__ void k_conv_big(const __bf16* __restrict__ xpad,
                           const __bf16* __restrict__ btiles,
                           const float*  __restrict__ impulse,
                           const float*  __restrict__ amp,
                           float*        __restrict__ out) {
    int wid  = blockIdx.x * 8 + (threadIdx.x >> 5);   // 16384 waves
    int lane = threadIdx.x & 31;
    int c  = wid >> 7;
    int t0 = (wid & 127) << 8;
    int h16 = lane >> 4, l15 = lane & 15;

    // A element offset: xpad[c][ t0 + 16*m + 16*kb + 8*h16 + e ]  (buffer incl. pad)
    const __bf16* xb = xpad + (size_t)c * XPAD_LEN + t0 + 16 * l15 + 8 * h16;
    const __bf16* bb = btiles + ((size_t)c * 256 * 32 + lane) * 16;

    v8f acc = {};
    Frag A, B;
    A.q[0] = *(const uint4*)(xb);                     // K elems 0..7 for kb=0
#pragma unroll 4
    for (int kb = 0; kb < 256; kb++) {
        A.q[1] = *(const uint4*)(xb + 16 + kb * 16);  // K elems 8..15 for this kb
        B.q[0] = *(const uint4*)(bb + kb * 512);
        B.q[1] = *(const uint4*)(bb + kb * 512 + 8);
        if (kb + 8 < 256) __builtin_prefetch(bb + (kb + 8) * 512, 0, 1);
        acc = __builtin_amdgcn_wmma_f32_16x16x32_bf16(false, A.v, false, B.v,
                                                      (short)0, acc, false, false);
        A.q[0] = A.q[1];                              // reuse high half as next low half
    }

    float a = amp[c];
#pragma unroll
    for (int r = 0; r < 8; r++) {
        int t = t0 + 16 * (r + 8 * h16) + l15;        // C/D layout: row = r + 8*h16
        float v = acc[r] + ((t < IMP_SAMP) ? impulse[c * IMP_SAMP + t] : 0.0f);
        out[(size_t)c * NSAMP + t] = v * a;
    }
}

// ---------------------------------------------------------------------------
// Launch
// ---------------------------------------------------------------------------
extern "C" void kernel_launch(void* const* d_in, const int* in_sizes, int n_in,
                              void* d_out, int out_size, void* d_ws, size_t ws_size,
                              hipStream_t stream) {
    const float* balance        = (const float*)d_in[0];  // 128x128
    const float* decay          = (const float*)d_in[1];  // 128x1
    const float* impulse_choice = (const float*)d_in[2];  // 128x16
    const float* filter_choice  = (const float*)d_in[3];  // 128x16
    const float* amp            = (const float*)d_in[4];  // 128
    const float* bank           = (const float*)d_in[5];  // 128x32768
    const float* filters        = (const float*)d_in[6];  // 16x512
    const float* impulses       = (const float*)d_in[7];  // 16x128
    float* out = (float*)d_out;

    // Workspace layout (all offsets 256B aligned)
    char* ws = (char*)d_ws;
    size_t o = 0;
    __bf16* xpad   = (__bf16*)(ws + o); o += (size_t)BATCH * XPAD_LEN * 2;        // 9.4 MB
    __bf16* btiles = (__bf16*)(ws + o); o += (size_t)BATCH * 256 * 32 * 16 * 2;   // 32 MB
    __bf16* abf    = (__bf16*)(ws + o); o += (size_t)BATCH * NFREQ * 2;           // 32 KB
    __bf16* bankT  = (__bf16*)(ws + o); o += (size_t)NSAMP * NFREQ * 2;           // 8 MB
    float*  imp_sig = (float*)(ws + o); o += (size_t)BATCH * IMP_SAMP * 4;        // 2 MB
    float*  filt    = (float*)(ws + o); o += (size_t)BATCH * FSIZE * 4;           // 256 KB
    float*  impulse = (float*)(ws + o); o += (size_t)BATCH * IMP_SAMP * 4;        // 2 MB

    // Stage 1: precision conversion / transpose / pad clearing
    k_bal  <<<64,   256, 0, stream>>>(balance, abf);
    k_bankT<<<4096, 256, 0, stream>>>(bank, bankT);
    k_zero <<<(BATCH * XPAD_LEN * 2 / 16) / 256, 256, 0, stream>>>((uint4*)xpad);

    // Stage 2: GEMM1 + fused window/decay weighting -> bf16 padded signal
    k_gemm_res<<<2048, 256, 0, stream>>>(abf, bankT, decay, xpad);

    // Stage 3: impulse synthesis chain
    k_impsig <<<2048, 256, 0, stream>>>(impulse_choice, impulses, imp_sig);
    k_filt   <<<256,  256, 0, stream>>>(filter_choice, filters, filt);
    k_impulse<<<2048, 256, 0, stream>>>(imp_sig, filt, impulse);
    k_btiles <<<4096, 256, 0, stream>>>(impulse, btiles);

    // Stage 4: the dominant 4096-tap grouped conv on WMMA, fused epilogue -> d_out
    k_conv_big<<<2048, 256, 0, stream>>>(xpad, btiles, impulse, amp, out);
}
